// GCNWithAggregator_79654463472119
// MI455X (gfx1250) — matrix operand order repaired and checked by reading, not compiled, not run on hardware
//
#include <hip/hip_runtime.h>
#include <hip/hip_bf16.h>

// ---------------------------------------------------------------------------
// CDNA5 (gfx1250) GCN pipeline.
// Dominant GEMM (25000x4096x1024) + all other matmuls run through
// v_wmma_f32_16x16x32_bf16 (bf16 inputs, fp32 accumulate).
// ---------------------------------------------------------------------------

typedef __attribute__((ext_vector_type(16))) __bf16 v16bf;
typedef __attribute__((ext_vector_type(8)))  float  v8f;

union ABFrag { v16bf v; uint4 q[2]; };

__device__ __forceinline__ unsigned int packbf(float x, float y) {
    union { __bf16 b[2]; unsigned int u; } t;
    t.b[0] = (__bf16)x; t.b[1] = (__bf16)y;
    return t.u;
}

#define BM 128
#define BN 128
#define KB 32
#define LDS_STRIDE 40   // bf16 elements per LDS row (32 data + 8 pad, keeps 16B align)

// C[M x N] = fuse(A[M x Kpad] (fp32) @ Bt[Npad x Kpad]^T (bf16) , bias)
// fuse: 0 = none, 1 = +bias, 2 = relu(+bias)
__global__ __launch_bounds__(256)
void gemm_bf16_wmma(const float* __restrict__ A,
                    const __bf16* __restrict__ Bt,
                    const float* __restrict__ bias,
                    float* __restrict__ C,
                    int M, int N, int Kpad, int ldc, int fuse)
{
    __shared__ __bf16 As[BM * LDS_STRIDE];
    __shared__ __bf16 Bs[BN * LDS_STRIDE];

    const int tid  = threadIdx.x;
    const int wid  = tid >> 5;
    const int lane = tid & 31;
    const int m0 = blockIdx.y * BM;
    const int n0 = blockIdx.x * BN;

    const int waveM = (wid & 3) * 32;   // 4 wave rows of 32
    const int waveN = (wid >> 2) * 64;  // 2 wave cols of 64

    v8f acc[2][4];
    const v8f vzero = {0.f,0.f,0.f,0.f,0.f,0.f,0.f,0.f};
    for (int i = 0; i < 2; ++i)
        for (int j = 0; j < 4; ++j)
            acc[i][j] = vzero;

    // --- global load mappings (256 threads) ---
    // A tile: 128 rows x 32 fp32; 2 threads/row, 16 contiguous fp32 each
    const int arow = tid >> 1;
    const int aseg = (tid & 1) * 16;
    long aRowG = (long)m0 + arow;
    if (aRowG >= M) aRowG = M - 1;             // clamp (results masked at store)
    const float* aptr = A + aRowG * (long)Kpad + aseg;

    // B tile: 128 rows x 32 bf16; 2 threads/row, 16 contiguous bf16 each
    const int brow = tid >> 1;
    const int bseg = (tid & 1) * 16;
    const __bf16* bptr = Bt + (long)(n0 + brow) * Kpad + bseg;

    const int ch = lane >> 4;   // lane half (K-half selector)
    const int lr = lane & 15;   // row/col within 16

    for (int k0 = 0; k0 < Kpad; k0 += KB) {
        __syncthreads();
        // stage A (fp32 -> bf16)
        {
            const float4* ap = reinterpret_cast<const float4*>(aptr + k0);
            float4 a0 = ap[0], a1 = ap[1], a2 = ap[2], a3 = ap[3];
            unsigned int* adst =
                reinterpret_cast<unsigned int*>(&As[arow * LDS_STRIDE + aseg]);
            adst[0] = packbf(a0.x, a0.y);
            adst[1] = packbf(a0.z, a0.w);
            adst[2] = packbf(a1.x, a1.y);
            adst[3] = packbf(a1.z, a1.w);
            adst[4] = packbf(a2.x, a2.y);
            adst[5] = packbf(a2.z, a2.w);
            adst[6] = packbf(a3.x, a3.y);
            adst[7] = packbf(a3.z, a3.w);
        }
        // stage B (already bf16, n-major)
        {
            const uint4* bp = reinterpret_cast<const uint4*>(bptr + k0);
            uint4 b0 = bp[0], b1 = bp[1];
            uint4* bdst = reinterpret_cast<uint4*>(&Bs[brow * LDS_STRIDE + bseg]);
            bdst[0] = b0;
            bdst[1] = b1;
        }
        __syncthreads();

        // build fragments per ISA 16-bit A/B striping
        ABFrag afrag[2], bfrag[4];
        #pragma unroll
        for (int i = 0; i < 2; ++i) {
            const __bf16* rp = &As[(waveM + i * 16 + lr) * LDS_STRIDE];
            afrag[i].q[0] = *reinterpret_cast<const uint4*>(rp + ch * 8);
            afrag[i].q[1] = *reinterpret_cast<const uint4*>(rp + 16 + ch * 8);
        }
        #pragma unroll
        for (int j = 0; j < 4; ++j) {
            const __bf16* rp = &Bs[(waveN + j * 16 + lr) * LDS_STRIDE];
            bfrag[j].q[0] = *reinterpret_cast<const uint4*>(rp + ch * 8);
            bfrag[j].q[1] = *reinterpret_cast<const uint4*>(rp + 16 + ch * 8);
        }
        #pragma unroll
        for (int i = 0; i < 2; ++i)
            #pragma unroll
            for (int j = 0; j < 4; ++j)
                acc[i][j] = __builtin_amdgcn_wmma_f32_16x16x32_bf16(
                    false, afrag[i].v, false, bfrag[j].v,
                    (short)0, acc[i][j], false, false);
    }

    // write-out: lane half ch owns rows c*8..c*8+7 of each 16x16 subtile
    #pragma unroll
    for (int j = 0; j < 4; ++j) {
        const int col = n0 + waveN + j * 16 + lr;
        if (col >= N) continue;
        const float bv = (fuse && bias) ? bias[col] : 0.f;
        #pragma unroll
        for (int i = 0; i < 2; ++i) {
            #pragma unroll
            for (int r = 0; r < 8; ++r) {
                const int row = m0 + waveM + i * 16 + ch * 8 + r;
                if (row < M) {
                    float v = acc[i][j][r];
                    if (fuse) { v += bv; if (fuse == 2 && v < 0.f) v = 0.f; }
                    C[(long)row * ldc + col] = v;
                }
            }
        }
    }
}

// ---- helper kernels --------------------------------------------------------

// W[K x Nout] fp32 -> Wt[Npad x Kpad] bf16 (transposed, zero-padded)
__global__ void convert_wt_kernel(const float* __restrict__ W, __bf16* __restrict__ Wt,
                                  int K, int Nout, int Kpad, long total)
{
    long idx = (long)blockIdx.x * 256 + threadIdx.x;
    if (idx >= total) return;
    int k = (int)(idx % Kpad);
    int n = (int)(idx / Kpad);
    float v = 0.f;
    if (n < Nout && k < K) v = W[(long)k * Nout + n];
    Wt[idx] = (__bf16)v;
}

// build xt[N x 224]: zeros everywhere; gene rows get x[row, 0:198]
__global__ void xt_init_kernel(const float* __restrict__ x, float* __restrict__ xt,
                               int NG, int FEAT, long total)
{
    long t = (long)blockIdx.x * 256 + threadIdx.x;
    if (t >= total) return;
    int row = (int)(t / 224);
    int col = (int)(t % 224);
    float v = 0.f;
    if (row < NG && col < 198) v = x[(long)row * FEAT + col];
    xt[t] = v;
}

__global__ void deg_init_kernel(float* deg, int n) {
    int i = blockIdx.x * 256 + threadIdx.x;
    if (i < n) deg[i] = 1.0f;   // self-loop contribution
}

__global__ void deg_accum_kernel(const int* __restrict__ dst, float* deg, int E) {
    int e = blockIdx.x * 256 + threadIdx.x;
    if (e < E) atomicAdd(&deg[dst[e]], 1.0f);
}

__global__ void rsqrt_kernel(float* d, int n) {
    int i = blockIdx.x * 256 + threadIdx.x;
    if (i < n) d[i] = rsqrtf(d[i]);
}

// agg[i][f] = hpre[i][f] * dinv[i]^2   (self-loop term + init)
__global__ void selfloop_kernel(const float* __restrict__ hpre,
                                const float* __restrict__ dinv,
                                float* __restrict__ agg, int F, long total)
{
    long t = (long)blockIdx.x * 256 + threadIdx.x;
    if (t >= total) return;
    int node = (int)(t / F);
    float w = dinv[node];
    agg[t] = hpre[t] * (w * w);
}

// scatter: agg[dst] += hpre[src] * dinv[src]*dinv[dst]  (64 threads/edge x4 floats)
__global__ void edge_scatter_kernel(const int* __restrict__ src,
                                    const int* __restrict__ dst,
                                    const float* __restrict__ dinv,
                                    const float* __restrict__ hpre,
                                    float* __restrict__ agg,
                                    int E, int F)
{
    long t = (long)blockIdx.x * 256 + threadIdx.x;
    long total = (long)E * 64;
    if (t >= total) return;
    int e = (int)(t >> 6);
    int f = (int)(t & 63) << 2;
    int s = src[e], d = dst[e];
    float w = dinv[s] * dinv[d];
    const float4 v = *reinterpret_cast<const float4*>(hpre + (long)s * F + f);
    float* out = agg + (long)d * F + f;
    atomicAdd(out + 0, v.x * w);
    atomicAdd(out + 1, v.y * w);
    atomicAdd(out + 2, v.z * w);
    atomicAdd(out + 3, v.w * w);
}

__global__ void bias_relu_kernel(float* __restrict__ h, const float* __restrict__ b,
                                 int F, long total)
{
    long t = (long)blockIdx.x * 256 + threadIdx.x;
    if (t >= total) return;
    float v = h[t] + b[(int)(t % F)];
    h[t] = v > 0.f ? v : 0.f;
}

// ---- launch ----------------------------------------------------------------

static inline long cdiv(long a, long b) { return (a + b - 1) / b; }

extern "C" void kernel_launch(void* const* d_in, const int* in_sizes, int n_in,
                              void* d_out, int out_size, void* d_ws, size_t ws_size,
                              hipStream_t stream) {
    const float* x   = (const float*)d_in[0];
    const float* Wg1 = (const float*)d_in[1];
    const float* bg1 = (const float*)d_in[2];
    const float* Wg2 = (const float*)d_in[3];
    const float* bg2 = (const float*)d_in[4];
    const float* W1  = (const float*)d_in[5];
    const float* b1  = (const float*)d_in[6];
    const float* W2  = (const float*)d_in[7];
    const float* b2  = (const float*)d_in[8];
    const float* Wf  = (const float*)d_in[9];
    const float* bf_ = (const float*)d_in[10];
    const int*   ei  = (const int*)d_in[11];

    const int Nn = 50000, NG = 25000, FEAT = 4096;
    const int GOH = 1024, GO = 198, H = 256, OUT = 198;
    const int GO_KP = 224;              // 198 padded to mult of 32
    const int E = in_sizes[11] / 2;
    const int* src = ei;
    const int* dst = ei + E;

    // ---- workspace carve-out (256B aligned) ----
    char* wsb = (char*)d_ws;
    size_t off = 0;
    auto take = [&](size_t bytes) -> char* {
        char* p = wsb + off;
        off = (off + bytes + 255) & ~(size_t)255;
        return p;
    };
    __bf16* Wg1t = (__bf16*)take((size_t)1024 * 4096 * 2);   // [1024][4096]
    __bf16* Wg2t = (__bf16*)take((size_t)256  * 1024 * 2);   // [256][1024]
    __bf16* W1t  = (__bf16*)take((size_t)256  * 224  * 2);   // [256][224]
    __bf16* W2t  = (__bf16*)take((size_t)256  * 256  * 2);   // [256][256]
    __bf16* Wft  = (__bf16*)take((size_t)256  * 256  * 2);   // [256][256]
    float*  dinv = (float*)take((size_t)Nn * 4);
    float*  xt   = (float*)take((size_t)Nn * GO_KP * 4);          // 44.8 MB
    float*  bufA = (float*)take((size_t)NG * GOH * 4);            // 102.4 MB (go_h / h1pre / h2pre)
    float*  bufB = (float*)take((size_t)Nn * H * 4);              // 51.2 MB (agg / h)
    (void)ws_size; (void)n_in;

    // ---- 0) convert + transpose + pad weights to bf16 ----
    {
        long t1 = (long)1024 * 4096;
        convert_wt_kernel<<<cdiv(t1, 256), 256, 0, stream>>>(Wg1, Wg1t, FEAT, GOH, 4096, t1);
        long t2 = (long)256 * 1024;
        convert_wt_kernel<<<cdiv(t2, 256), 256, 0, stream>>>(Wg2, Wg2t, GOH, GO, 1024, t2);
        long t3 = (long)256 * 224;
        convert_wt_kernel<<<cdiv(t3, 256), 256, 0, stream>>>(W1, W1t, GO, H, 224, t3);
        long t4 = (long)256 * 256;
        convert_wt_kernel<<<cdiv(t4, 256), 256, 0, stream>>>(W2, W2t, H, H, 256, t4);
        convert_wt_kernel<<<cdiv(t4, 256), 256, 0, stream>>>(Wf, Wft, H, OUT, 256, t4);
    }

    // ---- 1) GOID MLP layer 1: go_h = relu(x_goid @ Wg1 + bg1)  [25000 x 1024] ----
    {
        dim3 grid(1024 / BN, (unsigned)cdiv(NG, BM));
        gemm_bf16_wmma<<<grid, 256, 0, stream>>>(x + (long)NG * FEAT, Wg1t, bg1,
                                                 bufA, NG, GOH, 4096, GOH, 2);
    }

    // ---- 2) build xt (zeros + gene rows), then MLP layer 2 -> xt goid rows ----
    {
        long t = (long)Nn * GO_KP;
        xt_init_kernel<<<cdiv(t, 256), 256, 0, stream>>>(x, xt, NG, FEAT, t);
        dim3 grid(256 / BN, (unsigned)cdiv(NG, BM));
        gemm_bf16_wmma<<<grid, 256, 0, stream>>>(bufA, Wg2t, bg2,
                                                 xt + (long)NG * GO_KP,
                                                 NG, GO, 1024, GO_KP, 2);
    }

    // ---- 3) degree -> dinv ----
    {
        deg_init_kernel<<<cdiv(Nn, 256), 256, 0, stream>>>(dinv, Nn);
        deg_accum_kernel<<<cdiv(E, 256), 256, 0, stream>>>(dst, dinv, E);
        rsqrt_kernel<<<cdiv(Nn, 256), 256, 0, stream>>>(dinv, Nn);
    }

    // ---- 4) conv1: h1pre = xt @ W1 ; propagate ; relu(+b1) ----
    {
        dim3 grid(256 / BN, (unsigned)cdiv(Nn, BM));
        gemm_bf16_wmma<<<grid, 256, 0, stream>>>(xt, W1t, nullptr, bufA,
                                                 Nn, H, GO_KP, H, 0);
        long t = (long)Nn * H;
        selfloop_kernel<<<cdiv(t, 256), 256, 0, stream>>>(bufA, dinv, bufB, H, t);
        edge_scatter_kernel<<<cdiv((long)E * 64, 256), 256, 0, stream>>>(
            src, dst, dinv, bufA, bufB, E, H);
        bias_relu_kernel<<<cdiv(t, 256), 256, 0, stream>>>(bufB, b1, H, t);
    }

    // ---- 5) conv2: h2pre = h1 @ W2 ; propagate ; relu(+b2) ----
    {
        dim3 grid(256 / BN, (unsigned)cdiv(Nn, BM));
        gemm_bf16_wmma<<<grid, 256, 0, stream>>>(bufB, W2t, nullptr, bufA,
                                                 Nn, H, 256, H, 0);
        long t = (long)Nn * H;
        selfloop_kernel<<<cdiv(t, 256), 256, 0, stream>>>(bufA, dinv, bufB, H, t);
        edge_scatter_kernel<<<cdiv((long)E * 64, 256), 256, 0, stream>>>(
            src, dst, dinv, bufA, bufB, E, H);
        bias_relu_kernel<<<cdiv(t, 256), 256, 0, stream>>>(bufB, b2, H, t);
    }

    // ---- 6) final: out = h2 @ Wf + bf  [50000 x 198] ----
    {
        dim3 grid(256 / BN, (unsigned)cdiv(Nn, BM));
        gemm_bf16_wmma<<<grid, 256, 0, stream>>>(bufB, Wft, bf_, (float*)d_out,
                                                 Nn, OUT, 256, OUT, 1);
    }
    (void)out_size;
}
